// VQVAE_61555471286675
// MI455X (gfx1250) — matrix operand order repaired and checked by reading, not compiled, not run on hardware
//
#include <hip/hip_runtime.h>
#include <stdint.h>

// ---------------------------------------------------------------------------
// VQVAE forward for MI455X (gfx1250, wave32, WMMA).
// All GEMMs: bf16 x bf16 -> fp32 via v_wmma_f32_16x16x32_bf16.
// Staging: GLOBAL_LOAD_ASYNC_TO_LDS_B128 (ASYNCcnt) + double-buffered LDS.
// ---------------------------------------------------------------------------

typedef __bf16 bf16_t;
typedef __attribute__((ext_vector_type(16))) __bf16 v16bf;
typedef __attribute__((ext_vector_type(8)))  __bf16 v8bf;
typedef __attribute__((ext_vector_type(8)))  float  v8f;

#define N_ROWS 8192
#define DIM    1024
#define HID    4096
#define KCB    4096   // codebook size

// -------------------------- small utility kernels --------------------------

__global__ __launch_bounds__(256)
void cvt_f32_bf16(const float* __restrict__ in, bf16_t* __restrict__ out, int n) {
  int i = blockIdx.x * 256 + threadIdx.x;
  if (i < n) out[i] = (bf16_t)in[i];
}

__global__ __launch_bounds__(256)
void emb_sqnorm(const float* __restrict__ emb, float* __restrict__ e2) {
  __shared__ float red[256];
  int k = blockIdx.x;
  float s = 0.f;
  for (int d = threadIdx.x; d < DIM; d += 256) {
    float v = emb[(size_t)k * DIM + d];
    s += v * v;
  }
  red[threadIdx.x] = s;
  __syncthreads();
  for (int off = 128; off > 0; off >>= 1) {
    if (threadIdx.x < (unsigned)off) red[threadIdx.x] += red[threadIdx.x + off];
    __syncthreads();
  }
  if (threadIdx.x == 0) e2[k] = red[0];
}

__global__ __launch_bounds__(256)
void init_keys(unsigned long long* __restrict__ keys) {
  int i = blockIdx.x * 256 + threadIdx.x;
  if (i < N_ROWS) keys[i] = ~0ull;
}

// Mueller hash with torch/jax int64 wraparound semantics (arithmetic >>, mul wraps)
__device__ __forceinline__ long long mueller_hash(long long x) {
  x = (x >> 16) ^ x;
  x = (long long)((unsigned long long)x * 73244475ull);
  x = (x >> 16) ^ x;
  x = (long long)((unsigned long long)x * 73244475ull);
  x = (x >> 16) ^ x;
  return x;
}

__global__ __launch_bounds__(256)
void finalize_k(const unsigned long long* __restrict__ keys,
                const float* __restrict__ emb, float* __restrict__ out) {
  int n = blockIdx.x;
  long long idx = (long long)(unsigned int)(keys[n] & 0xFFFFFFFFull);
  int sel0 = (int)(mueller_hash(idx + (long long)(1 * KCB)) & (long long)(KCB - 1));
  int sel1 = (int)(mueller_hash(idx + (long long)(2 * KCB)) & (long long)(KCB - 1));
  int sel2 = (int)(mueller_hash(idx + (long long)(3 * KCB)) & (long long)(KCB - 1));
  for (int d = threadIdx.x; d < DIM; d += 256) {
    float s = emb[(size_t)sel0 * DIM + d] / 3.0f
            + emb[(size_t)sel1 * DIM + d] / 3.0f
            + emb[(size_t)sel2 * DIM + d] / 3.0f;
    out[(size_t)n * DIM + d] = s;
  }
}

// --------------------- CDNA5 async global->LDS copy ------------------------
// One instruction moves 16B per active lane straight into LDS, tracked with
// ASYNCcnt (no VGPR round trip, no loadcnt pressure).
__device__ __forceinline__ void async_copy_b128(unsigned lds_off, const bf16_t* gptr) {
  asm volatile("global_load_async_to_lds_b128 %0, %1, off"
               :: "v"(lds_off), "v"(gptr) : "memory");
}
__device__ __forceinline__ void wait_async_le6() {
  asm volatile("s_wait_asynccnt 0x6" ::: "memory");
}
__device__ __forceinline__ void wait_async_0() {
  asm volatile("s_wait_asynccnt 0x0" ::: "memory");
}

// ------------------------------- WMMA GEMM ---------------------------------
// C[N,M] = A[N,K] @ B[M,K]^T  (B row-major = weight matrix, like x @ W.T)
// Block tile 256x128, BK=32, 256 threads = 8 waves in a 4(M) x 2(N) grid;
// each wave computes a 64x64 region = 4x4 tiles of 16x16:
//   16 wmma per K-step vs 16 ds_load_b128 fragment reads (1.0 ratio).
// Double-buffered LDS fed by global_load_async_to_lds_b128.
// MODE: 0 = +bias, store bf16
//       1 = +bias, fourier (sin even col / cos odd col), store bf16
//       2 = +bias, store [h | fourier(h)] into width-2M output (layer 1 concat)
//       3 = argmin epilogue: key = pack(e2[col] - 2*score, col), atomicMin per row
#define LDS_STRIDE 40   // 32 + 8 bf16 pad -> 20 dwords/row, conflict-free b128 reads

template <int MODE>
__global__ __launch_bounds__(256)
void gemm_bf16(const bf16_t* __restrict__ A, const bf16_t* __restrict__ B,
               const float* __restrict__ bias, bf16_t* __restrict__ out,
               int K, int M, int outW,
               const float* __restrict__ e2,
               unsigned long long* __restrict__ keys) {
  __shared__ __align__(16) bf16_t sA[2][256 * LDS_STRIDE];
  __shared__ __align__(16) bf16_t sB[2][128 * LDS_STRIDE];

  const int t     = threadIdx.x;
  const int lane  = t & 31;
  const int wv    = t >> 5;
  const int waveM = (wv >> 1) * 64;   // 0,64,128,192
  const int waveN = (wv & 1) * 64;    // 0,64
  const int rowBaseA = blockIdx.y * 256;
  const int rowBaseB = blockIdx.x * 128;

  v8f acc[4][4];
#pragma unroll
  for (int i = 0; i < 4; i++)
#pragma unroll
    for (int j = 0; j < 4; j++)
      acc[i][j] = (v8f){0.f, 0.f, 0.f, 0.f, 0.f, 0.f, 0.f, 0.f};

  const int seg = t & 3;   // 16B segment of the 32-wide K slab
  const int row = t >> 2;  // 0..63

  // per-iteration staging: 4 async b128 for A (rows row+{0,64,128,192}),
  // 2 for B (rows row+{0,64}) -> ASYNCcnt += 6 per step per wave
  auto stage = [&](int buf, int k0) {
#pragma unroll
    for (int r = 0; r < 4; r++) {
      unsigned lo = (unsigned)(uintptr_t)
          (&sA[buf][(size_t)(row + 64 * r) * LDS_STRIDE + seg * 8]);
      async_copy_b128(lo, A + (size_t)(rowBaseA + row + 64 * r) * K + k0 + seg * 8);
    }
#pragma unroll
    for (int r = 0; r < 2; r++) {
      unsigned lo = (unsigned)(uintptr_t)
          (&sB[buf][(size_t)(row + 64 * r) * LDS_STRIDE + seg * 8]);
      async_copy_b128(lo, B + (size_t)(rowBaseB + row + 64 * r) * K + k0 + seg * 8);
    }
  };

  stage(0, 0);  // prologue

  const int nsteps = K >> 5;
  for (int s = 0; s < nsteps; s++) {
    const int cur = s & 1;
    if (s + 1 < nsteps) {
      stage(cur ^ 1, (s + 1) * 32);  // prefetch next tile into the other buffer
      wait_async_le6();              // in-order: tile s's 6 asyncs now complete
    } else {
      wait_async_0();
    }
    __syncthreads();

    // ---- B fragments (32x16 bf16): lane = column; lanes 0-15 K=0..15,
    //      lanes 16-31 K=16..31 (two consecutive b128 reads)
    v16bf bfrag[4];
    {
      const int n16  = lane & 15;
      const int koff = (lane < 16) ? 0 : 16;
#pragma unroll
      for (int j = 0; j < 4; j++) {
        const bf16_t* p = &sB[cur][(size_t)(waveN + 16 * j + n16) * LDS_STRIDE + koff];
        union { v16bf v; v8bf h[2]; } u;
        u.h[0] = *(const v8bf*)(p);
        u.h[1] = *(const v8bf*)(p + 8);
        bfrag[j] = u.v;
      }
    }

    // ---- A fragments (16x32 bf16): lane = row; lane<16 -> K{0..15},
    //      lane>=16 -> K{16..31} (two b128 reads)
    {
      const int m16 = lane & 15;
      const int kb  = (lane < 16) ? 0 : 16;
#pragma unroll
      for (int i = 0; i < 4; i++) {
        const bf16_t* p = &sA[cur][(size_t)(waveM + 16 * i + m16) * LDS_STRIDE];
        union { v16bf v; v8bf h[2]; } u;
        u.h[0] = *(const v8bf*)(p + kb);
        u.h[1] = *(const v8bf*)(p + 8 + kb);
        v16bf afrag = u.v;
#pragma unroll
        for (int j = 0; j < 4; j++) {
          acc[i][j] = __builtin_amdgcn_wmma_f32_16x16x32_bf16(
              false, afrag, false, bfrag[j], (short)0, acc[i][j], false, false);
        }
      }
    }
    __syncthreads();  // all waves done reading buf[cur] before it is re-staged
  }

  // ---- epilogue ----
  // C/D layout: lane holds col n = lane&15 ; VGPR v holds row m = v + (lane<16?0:8)
  const int mrow0 = rowBaseA + waveM + ((lane >> 4) << 3);
  const int ncol0 = rowBaseB + waveN + (lane & 15);

#pragma unroll
  for (int i = 0; i < 4; i++) {
#pragma unroll
    for (int j = 0; j < 4; j++) {
      const int gc = ncol0 + 16 * j;
      if (MODE == 3) {
        const float ee = e2[gc];
#pragma unroll
        for (int v = 0; v < 8; v++) {
          const int gr = mrow0 + 16 * i + v;
          float d = ee - 2.0f * acc[i][j][v];   // row-constant |z|^2 dropped
          unsigned int ub = __float_as_uint(d);
          ub = (ub & 0x80000000u) ? ~ub : (ub | 0x80000000u);  // sortable float
          unsigned long long key =
              ((unsigned long long)ub << 32) | (unsigned int)gc;
          // reduce over the 16 lanes that share this row (tie -> lowest idx)
          unsigned long long o;
          o = __shfl_xor(key, 1, 32); key = (o < key) ? o : key;
          o = __shfl_xor(key, 2, 32); key = (o < key) ? o : key;
          o = __shfl_xor(key, 4, 32); key = (o < key) ? o : key;
          o = __shfl_xor(key, 8, 32); key = (o < key) ? o : key;
          if ((lane & 15) == 0) atomicMin(keys + gr, key);
        }
      } else {
        const float bv = bias[gc];
#pragma unroll
        for (int v = 0; v < 8; v++) {
          const int gr = mrow0 + 16 * i + v;
          float val = acc[i][j][v] + bv;
          if (MODE == 0) {
            out[(size_t)gr * outW + gc] = (bf16_t)val;
          } else if (MODE == 1) {
            float f = (gc & 1) ? cosf(val) : sinf(val);
            out[(size_t)gr * outW + gc] = (bf16_t)f;
          } else {  // MODE == 2 : concat [h | fourier(h)]
            out[(size_t)gr * outW + gc] = (bf16_t)val;
            float f = (gc & 1) ? cosf(val) : sinf(val);
            out[(size_t)gr * outW + M + gc] = (bf16_t)f;
          }
        }
      }
    }
  }
}

// ------------------------------- launcher ----------------------------------

extern "C" void kernel_launch(void* const* d_in, const int* in_sizes, int n_in,
                              void* d_out, int out_size, void* d_ws, size_t ws_size,
                              hipStream_t stream) {
  (void)in_sizes; (void)n_in; (void)out_size; (void)ws_size;

  const float* x   = (const float*)d_in[0];
  const float* w1  = (const float*)d_in[1];
  const float* b1  = (const float*)d_in[2];
  const float* w2  = (const float*)d_in[3];
  const float* b2  = (const float*)d_in[4];
  const float* w3  = (const float*)d_in[5];
  const float* b3  = (const float*)d_in[6];
  const float* w4  = (const float*)d_in[7];
  const float* b4  = (const float*)d_in[8];
  const float* emb = (const float*)d_in[9];
  float* out = (float*)d_out;

  char* ws = (char*)d_ws;
  size_t off = 0;
  auto take = [&](size_t bytes) -> char* {
    char* p = ws + off;
    off += (bytes + 255) & ~(size_t)255;
    return p;
  };
  bf16_t* xb   = (bf16_t*)take((size_t)N_ROWS * DIM * 2);
  bf16_t* w1b  = (bf16_t*)take((size_t)HID * DIM * 2);
  bf16_t* w2b  = (bf16_t*)take((size_t)HID * 2 * HID * 2);
  bf16_t* w3b  = (bf16_t*)take((size_t)HID * HID * 2);
  bf16_t* w4b  = (bf16_t*)take((size_t)DIM * HID * 2);
  bf16_t* embb = (bf16_t*)take((size_t)KCB * DIM * 2);
  bf16_t* h1   = (bf16_t*)take((size_t)N_ROWS * 2 * HID * 2);
  bf16_t* h2   = (bf16_t*)take((size_t)N_ROWS * HID * 2);
  bf16_t* h3   = (bf16_t*)take((size_t)N_ROWS * HID * 2);
  bf16_t* zb   = (bf16_t*)take((size_t)N_ROWS * DIM * 2);
  float*  e2   = (float*)take((size_t)KCB * 4);
  unsigned long long* keys = (unsigned long long*)take((size_t)N_ROWS * 8);

  dim3 blk(256);
  auto cvt = [&](const float* src, bf16_t* dst, size_t n) {
    cvt_f32_bf16<<<dim3((unsigned)((n + 255) / 256)), blk, 0, stream>>>(src, dst, (int)n);
  };
  cvt(x,   xb,   (size_t)N_ROWS * DIM);
  cvt(w1,  w1b,  (size_t)HID * DIM);
  cvt(w2,  w2b,  (size_t)HID * 2 * HID);
  cvt(w3,  w3b,  (size_t)HID * HID);
  cvt(w4,  w4b,  (size_t)DIM * HID);
  cvt(emb, embb, (size_t)KCB * DIM);

  emb_sqnorm<<<dim3(KCB), blk, 0, stream>>>(emb, e2);
  init_keys<<<dim3((N_ROWS + 255) / 256), blk, 0, stream>>>(keys);

  // L1: h1 = [x@w1.T + b1 | fourier(.)]  -> [8192, 8192] bf16
  gemm_bf16<2><<<dim3(HID / 128, N_ROWS / 256), blk, 0, stream>>>(
      xb, w1b, b1, h1, DIM, HID, 2 * HID, nullptr, nullptr);
  // L2: h2 = fourier(h1@w2.T + b2)       -> [8192, 4096] bf16
  gemm_bf16<1><<<dim3(HID / 128, N_ROWS / 256), blk, 0, stream>>>(
      h1, w2b, b2, h2, 2 * HID, HID, HID, nullptr, nullptr);
  // L3: h3 = fourier(h2@w3.T + b3)       -> [8192, 4096] bf16
  gemm_bf16<1><<<dim3(HID / 128, N_ROWS / 256), blk, 0, stream>>>(
      h2, w3b, b3, h3, HID, HID, HID, nullptr, nullptr);
  // L4: z = h3@w4.T + b4                 -> [8192, 1024] bf16
  gemm_bf16<0><<<dim3(DIM / 128, N_ROWS / 256), blk, 0, stream>>>(
      h3, w4b, b4, zb, HID, DIM, DIM, nullptr, nullptr);
  // L5: scores = z@emb.T ; fused argmin over e2[k]-2*score -> keys
  gemm_bf16<3><<<dim3(KCB / 128, N_ROWS / 256), blk, 0, stream>>>(
      zb, embb, nullptr, nullptr, DIM, KCB, KCB, e2, keys);

  // gather: out[n] = sum_{i=1..3} emb[hash(idx+i*K) & (K-1)] / 3
  finalize_k<<<dim3(N_ROWS), blk, 0, stream>>>(keys, emb, out);
}
